// SADGC_41807211659526
// MI455X (gfx1250) — matrix operand order repaired and checked by reading, not compiled, not run on hardware
//
#include <hip/hip_runtime.h>
#include <hip/hip_fp16.h>

typedef __attribute__((ext_vector_type(16))) _Float16 v16h;
typedef __attribute__((ext_vector_type(8)))  float    v8f;

#define B_  16
#define N_  512
#define T_  32
#define C_  128
#define R_  32
#define D_  64

__device__ __forceinline__ float sigmoidf_(float x) { return 1.0f / (1.0f + __expf(-x)); }

// ------------------------------------------------------------------
// Stage 1: per-(b,n) gating + temporal softmax pool + Q/K projection.
// One 128-thread block per (b,n). Outputs Q,K as f16 for WMMA.
// ------------------------------------------------------------------
__global__ __launch_bounds__(128) void stage1_kernel(
    const float* __restrict__ x, const float* __restrict__ Ws,
    const float* __restrict__ Wphi, const float* __restrict__ b_phi,
    const float* __restrict__ Wmu, const float* __restrict__ b_mu,
    const float* __restrict__ WQ, const float* __restrict__ WK,
    _Float16* __restrict__ Qh, _Float16* __restrict__ Kh)
{
    __shared__ float sWsT[C_ * R_];   // Ws transposed: [c][r] (bank-conflict free dots)
    __shared__ float sXhat[T_ * C_];  // [t][c]
    __shared__ float sAvg[C_];
    __shared__ float sZca[R_];
    __shared__ float sZta[T_ * R_];   // relu'd
    __shared__ float sAlpha[T_];
    __shared__ float sXc[C_];

    const int tid = threadIdx.x;            // == channel c
    const size_t bn = blockIdx.x;
    const float* xp = x + bn * (size_t)(T_ * C_);

    // Ws (r,C) row-major -> LDS transposed
    for (int i = 0; i < (R_ * C_) / 128; ++i) {
        int idx = tid + i * 128;
        int r = idx >> 7;          // / C_
        int c = idx & (C_ - 1);
        sWsT[c * R_ + r] = Ws[idx];
    }

    // load my channel across T, accumulate mean
    float xt[T_];
    float acc = 0.f;
    #pragma unroll
    for (int t = 0; t < T_; ++t) { xt[t] = xp[t * C_ + tid]; acc += xt[t]; }
    sAvg[tid] = acc * (1.0f / T_);
    __syncthreads();

    // z_ca = relu(Ws @ avg)   (threads 0..31, one r each)
    if (tid < R_) {
        float s = 0.f;
        for (int c = 0; c < C_; ++c) s += sWsT[c * R_ + tid] * sAvg[c];
        sZca[tid] = s > 0.f ? s : 0.f;
    }
    __syncthreads();

    // w[c] = sigmoid(Wphi[c,:] @ z_ca + b_phi[c]);  X_hat = X_w * w
    {
        float s = 0.f;
        const float* wp = Wphi + tid * R_;
        for (int r = 0; r < R_; ++r) s += wp[r] * sZca[r];
        float wv = sigmoidf_(s + b_phi[tid]);
        #pragma unroll
        for (int t = 0; t < T_; ++t) { xt[t] *= wv; sXhat[t * C_ + tid] = xt[t]; }
    }
    __syncthreads();

    // z_ta[t][r] = relu(X_hat[t,:] @ Ws[r,:]) : 1024 dots of length 128, 8/thread.
    // Mapping q = tid + 128*i -> each wave-iteration covers one t with r = lane
    // (sXhat broadcast, sWsT conflict-free).
    for (int i = 0; i < 8; ++i) {
        int q = tid + 128 * i;
        int t = q >> 5;
        int r = q & 31;
        float a = 0.f;
        for (int c = 0; c < C_; ++c) a += sXhat[t * C_ + c] * sWsT[c * R_ + r];
        sZta[t * R_ + r] = a > 0.f ? a : 0.f;
    }
    __syncthreads();

    // logits + softmax over T on wave 0 (one t per lane, wave32 reductions)
    if (tid < T_) {
        float l = b_mu[0];
        for (int r = 0; r < R_; ++r) l += sZta[tid * R_ + r] * Wmu[r];
        float m = l;
        for (int off = 16; off > 0; off >>= 1) m = fmaxf(m, __shfl_xor(m, off, 32));
        float e = __expf(l - m);
        float tot = e;
        for (int off = 16; off > 0; off >>= 1) tot += __shfl_xor(tot, off, 32);
        sAlpha[tid] = e / tot;
    }
    __syncthreads();

    // X_c[c] = sum_t alpha[t] * X_hat[t][c]
    {
        float xc = 0.f;
        #pragma unroll
        for (int t = 0; t < T_; ++t) xc += sAlpha[t] * sXhat[t * C_ + tid];
        sXc[tid] = xc;
    }
    __syncthreads();

    // Q (threads 0..63) / K (threads 64..127): 64 dots of 128, emit f16
    {
        int dq = tid & 63;
        const float* Wrow = (tid < 64 ? WQ : WK) + dq * C_;
        float q = 0.f;
        for (int c = 0; c < C_; ++c) q += Wrow[c] * sXc[c];
        _Float16* dst = (tid < 64 ? Qh : Kh) + bn * D_ + dq;
        *dst = (_Float16)q;
    }
}

// ------------------------------------------------------------------
// Stage 2: E[b] = (Q[b] @ K[b]^T) / 8  via v_wmma_f32_16x16x32_f16.
// One wave per (b, n-tile, m-group); each wave does 8 m-tiles (A reuse).
// Fragment layouts follow ISA 7.12.2 (16-bit A 16x32; B 32x16; f32 C/D).
// ------------------------------------------------------------------
__global__ __launch_bounds__(256) void gemm_kernel(
    const _Float16* __restrict__ Qh, const _Float16* __restrict__ Kh,
    float* __restrict__ E)
{
    const int tid  = threadIdx.x;
    const int lane = tid & 31;
    const int wid  = blockIdx.x * (blockDim.x >> 5) + (tid >> 5);

    const int b      = wid >> 7;      // / (32 n-tiles * 4 m-groups)
    const int rem    = wid & 127;
    const int ntile  = rem >> 2;
    const int mgroup = rem & 3;
    const int n0     = ntile * 16;

    const int half = lane >> 4;       // lane half selects K sub-range
    const int lrow = lane & 15;

    // A fragments: Q rows n0+lrow, K-chunks [0,32) and [32,64).
    // lane<16: v0..3 = K pairs {0,2,4,6}+.. ; v4..7 = {16,18,20,22}+..
    // lane>=16: +8 on each K base.
    union { v16h h; unsigned u[8]; } a0, a1;
    {
        const unsigned* qrow = (const unsigned*)(Qh + ((size_t)b * N_ + n0 + lrow) * D_);
        #pragma unroll
        for (int p = 0; p < 8; ++p) {
            int kb = ((p < 4) ? 2 * p : 16 + 2 * (p - 4)) + 8 * half;  // in halves
            a0.u[p] = qrow[kb >> 1];
            a1.u[p] = qrow[(kb + 32) >> 1];
        }
    }

    for (int mi = 0; mi < 8; ++mi) {
        const int m0 = (mgroup * 8 + mi) * 16;
        // B fragment = K^T tile: lane holds column m0+lrow;
        // same K order as A: lane<16: {0..7,16..23}; lane>=16: +8.
        union { v16h h; unsigned u[8]; } b0, b1;
        const unsigned* krow = (const unsigned*)(Kh + ((size_t)b * N_ + m0 + lrow) * D_);
        #pragma unroll
        for (int p = 0; p < 8; ++p) {
            int kb = ((p < 4) ? 2 * p : 16 + 2 * (p - 4)) + 8 * half;
            b0.u[p] = krow[kb >> 1];
            b1.u[p] = krow[(kb + 32) >> 1];
        }
        v8f c = {};
        c = __builtin_amdgcn_wmma_f32_16x16x32_f16(false, a0.h, false, b0.h, (short)0, c, false, false);
        c = __builtin_amdgcn_wmma_f32_16x16x32_f16(false, a1.h, false, b1.h, (short)0, c, false, false);

        // C/D layout: VGPR i -> M = i + 8*half, N = lane&15
        #pragma unroll
        for (int i = 0; i < 8; ++i) {
            int row = n0 + i + 8 * half;
            E[((size_t)b * N_ + row) * N_ + m0 + lrow] = 0.125f * c[i];  // 1/sqrt(64)
        }
    }
}

// ------------------------------------------------------------------
// Stage 3: exact row-wise top-8 masking of sigmoid(E) (sigmoid monotone
// -> threshold in E-domain). One wave32 per row, 16 values/lane.
// transposed=1 reads columns of E (= rows of E^T) for S_b.
// ------------------------------------------------------------------
__global__ __launch_bounds__(256) void topk_kernel(
    const float* __restrict__ E, float* __restrict__ out, int transposed)
{
    const int lane = threadIdx.x & 31;
    const int row  = blockIdx.x * (blockDim.x >> 5) + (threadIdx.x >> 5);
    const int b = row >> 9;
    const int n = row & (N_ - 1);
    const float* Eb = E + (size_t)b * N_ * N_;

    float orig[16], work[16];
    #pragma unroll
    for (int j = 0; j < 16; ++j) {
        int m = lane + 32 * j;
        float v = transposed ? Eb[(size_t)m * N_ + n] : Eb[(size_t)n * N_ + m];
        orig[j] = v;
        work[j] = v;
    }

    // 8 rounds: wave-wide max, then exactly one instance removed
    // (ballot leader) -> exact 8th-largest including duplicates.
    float thr = 0.f;
    for (int it = 0; it < 8; ++it) {
        float lm = -3.4e38f;
        int   li = 0;
        #pragma unroll
        for (int j = 0; j < 16; ++j) if (work[j] > lm) { lm = work[j]; li = j; }
        float wm = lm;
        for (int off = 16; off > 0; off >>= 1) wm = fmaxf(wm, __shfl_xor(wm, off, 32));
        unsigned long long msk = __ballot(lm == wm);
        int leader = __ffsll(msk) - 1;
        if (lane == leader) work[li] = -3.4e38f;
        thr = wm;
    }

    float* orow = out + ((size_t)b * N_ + n) * N_;
    #pragma unroll
    for (int j = 0; j < 16; ++j) {
        int m = lane + 32 * j;
        float v = orig[j];
        orow[m] = (v >= thr) ? sigmoidf_(v) : 0.f;
    }
}

extern "C" void kernel_launch(void* const* d_in, const int* in_sizes, int n_in,
                              void* d_out, int out_size, void* d_ws, size_t ws_size,
                              hipStream_t stream)
{
    (void)in_sizes; (void)n_in; (void)out_size; (void)ws_size;

    const float* x     = (const float*)d_in[0];
    const float* Ws    = (const float*)d_in[1];
    const float* Wphi  = (const float*)d_in[2];
    const float* b_phi = (const float*)d_in[3];
    const float* Wmu   = (const float*)d_in[4];
    const float* b_mu  = (const float*)d_in[5];
    const float* WQ    = (const float*)d_in[6];
    const float* WK    = (const float*)d_in[7];

    char* ws = (char*)d_ws;
    _Float16* Qh = (_Float16*)ws;                                   // 1 MB
    _Float16* Kh = (_Float16*)(ws + (size_t)B_ * N_ * D_ * 2);      // 1 MB
    float*    E  = (float*)(ws + 2 * (size_t)B_ * N_ * D_ * 2);     // 16 MB

    float* S_f = (float*)d_out;
    float* S_b = S_f + (size_t)B_ * N_ * N_;

    stage1_kernel<<<B_ * N_, 128, 0, stream>>>(x, Ws, Wphi, b_phi, Wmu, b_mu, WQ, WK, Qh, Kh);

    // waves = B * (N/16 n-tiles) * 4 m-groups = 2048 ; 8 waves / block
    gemm_kernel<<<(B_ * (N_ / 16) * 4) / 8, 256, 0, stream>>>(Qh, Kh, E);

    // one wave per row: B*N rows, 8 waves / block
    topk_kernel<<<(B_ * N_) / 8, 256, 0, stream>>>(E, S_f, 0);
    topk_kernel<<<(B_ * N_) / 8, 256, 0, stream>>>(E, S_b, 1);
}
